// PrototypeLayer_38826504356175
// MI455X (gfx1250) — matrix operand order repaired and checked by reading, not compiled, not run on hardware
//
#include <hip/hip_runtime.h>
#include <hip/hip_bf16.h>

#define Bsz 256
#define Psz 512
#define Csz 512
#define Tsz 56
#define EPSV 1e-8f

typedef __attribute__((ext_vector_type(8)))  __bf16 v8bf;
typedef __attribute__((ext_vector_type(16))) __bf16 v16bf;
typedef __attribute__((ext_vector_type(8)))  float  v8f;
typedef __attribute__((ext_vector_type(4)))  float  f4;

// ---------------------------------------------------------------------------
// Prep: src (R,C,T) fp32  ->  dstA (T,R,C) bf16, optional dstY = w[c]*src,
// plus fp32 channel norms normT (T,R). LDS-tiled transpose, coalesced both ways.
// ---------------------------------------------------------------------------
__global__ __launch_bounds__(256) void prep_kernel(
    const float* __restrict__ src, __bf16* __restrict__ dstA,
    __bf16* __restrict__ dstY, const float* __restrict__ w,
    float* __restrict__ normT, int R)
{
    constexpr int CCH = 128;                      // channel chunk
    __shared__ float lds[CCH * Tsz];              // 28 KB
    __shared__ float nacc[Tsz];
    __shared__ float wsh[CCH];
    const int r = blockIdx.x;
    const float* s = src + (size_t)r * Csz * Tsz;

    for (int t = threadIdx.x; t < Tsz; t += blockDim.x) nacc[t] = 0.f;

    for (int c0 = 0; c0 < Csz; c0 += CCH) {
        __syncthreads();
        for (int i = threadIdx.x; i < CCH * Tsz; i += blockDim.x)
            lds[i] = s[(size_t)c0 * Tsz + i];     // contiguous read
        if (w != nullptr && threadIdx.x < CCH)
            wsh[threadIdx.x] = w[c0 + threadIdx.x];
        __syncthreads();
        // partial norms (fp32, exact data)
        for (int t = threadIdx.x; t < Tsz; t += blockDim.x) {
            float a = 0.f;
            for (int cc = 0; cc < CCH; ++cc) { float v = lds[cc * Tsz + t]; a += v * v; }
            nacc[t] += a;
        }
        // transposed bf16 writes: contiguous in c
        for (int i = threadIdx.x; i < CCH * Tsz; i += blockDim.x) {
            int t = i / CCH, cc = i % CCH;
            float v = lds[cc * Tsz + t];
            size_t o = ((size_t)t * R + r) * Csz + (c0 + cc);
            dstA[o] = (__bf16)v;
            if (dstY != nullptr) dstY[o] = (__bf16)(v * wsh[cc]);
        }
    }
    __syncthreads();
    for (int t = threadIdx.x; t < Tsz; t += blockDim.x)
        normT[(size_t)t * R + r] = sqrtf(nacc[t]);
}

// ---------------------------------------------------------------------------
// Tile loaders per ISA 7.12.2 bf16 layouts (rows are contiguous-C in the
// transposed buffers, so each chunk is one aligned b128 load).
// A (16x32, M x K): lane M = lane&15; lo half K={0..7,16..23}, hi K={8..15,24..31}
// B (32x16, K x N): lane N = lane&15; lo half K={0..7,16..23}, hi K={8..15,24..31}
// ---------------------------------------------------------------------------
__device__ __forceinline__ v16bf load_a_tile(const __bf16* p, int hi) {
    v8bf c0 = *(const v8bf*)(p + 8 * hi);
    v8bf c1 = *(const v8bf*)(p + 16 + 8 * hi);
    return __builtin_shufflevector(c0, c1, 0,1,2,3,4,5,6,7,8,9,10,11,12,13,14,15);
}
__device__ __forceinline__ v16bf load_b_tile(const __bf16* p, int hi) {
    v8bf c0 = *(const v8bf*)(p + 8 * hi);
    v8bf c1 = *(const v8bf*)(p + 16 + 8 * hi);
    return __builtin_shufflevector(c0, c1, 0,1,2,3,4,5,6,7,8,9,10,11,12,13,14,15);
}

// ---------------------------------------------------------------------------
// Fused dual-GEMM + online softmax + weighted reduce + fc/relu.
// Wave -> one 16(b) x 16(p) tile; block = 8 waves covering 16b x 128p
// (all 8 waves share the same A rows -> 8x WGP$ reuse of x/y tiles).
// ---------------------------------------------------------------------------
__global__ __launch_bounds__(256) void proto_wmma_kernel(
    const __bf16* __restrict__ xT, const __bf16* __restrict__ yT,
    const __bf16* __restrict__ pT, const float* __restrict__ xnT,
    const float* __restrict__ pnT, const float* __restrict__ fc_b,
    float* __restrict__ out)
{
    const int lane = threadIdx.x & 31;
    const int wave = threadIdx.x >> 5;
    const int hi   = lane >> 4;
    const int ln   = lane & 15;
    const int b0   = blockIdx.x * 16;
    const int p0   = blockIdx.y * 128 + wave * 16;

    v8f m, s, acc;
#pragma unroll
    for (int v = 0; v < 8; ++v) { m[v] = -3.4e38f; s[v] = 0.f; acc[v] = 0.f; }

    for (int t = 0; t < Tsz; ++t) {
        const __bf16* xr = xT + ((size_t)t * Bsz + b0 + ln) * Csz;
        const __bf16* yr = yT + ((size_t)t * Bsz + b0 + ln) * Csz;
        const __bf16* pr = pT + ((size_t)t * Psz + p0 + ln) * Csz;

        // warm next t's proto row (the only streaming operand; A rows are
        // block-shared and L0-resident). Lowers to global_prefetch_b8.
        __builtin_prefetch(pr + (size_t)Psz * Csz, 0, 1);

        v8f d = {0.f,0.f,0.f,0.f,0.f,0.f,0.f,0.f};
        v8f g = {0.f,0.f,0.f,0.f,0.f,0.f,0.f,0.f};
#pragma unroll 8
        for (int k = 0; k < Csz; k += 32) {
            v16bf a0 = load_a_tile(xr + k, hi);
            v16bf a1 = load_a_tile(yr + k, hi);
            v16bf bt = load_b_tile(pr + k, hi);
            d = __builtin_amdgcn_wmma_f32_16x16x32_bf16(false, a0, false, bt,
                                                        (short)0, d, false, false);
            g = __builtin_amdgcn_wmma_f32_16x16x32_bf16(false, a1, false, bt,
                                                        (short)0, g, false, false);
        }

        // normalization + online softmax update (C/D layout: M = v + 8*hi, N = ln)
        const float* xp = xnT + (size_t)t * Bsz + b0 + 8 * hi;
        f4 xa = *(const f4*)(xp);
        f4 xb = *(const f4*)(xp + 4);
        float pn = pnT[(size_t)t * Psz + p0 + ln];
        float xn8[8] = { xa.x, xa.y, xa.z, xa.w, xb.x, xb.y, xb.z, xb.w };
#pragma unroll
        for (int v = 0; v < 8; ++v) {
            float sim = d[v] * (1.0f / fmaxf(xn8[v] * pn, EPSV));
            float mn  = fmaxf(m[v], sim);
            float e0  = __expf(m[v] - mn);
            float e1  = __expf(sim  - mn);
            s[v]   = s[v]   * e0 + e1;
            acc[v] = acc[v] * e0 + e1 * g[v];
            m[v]   = mn;
        }
    }

    const float bias = fc_b[0];
#pragma unroll
    for (int v = 0; v < 8; ++v) {
        float o = fmaxf(acc[v] / s[v] + bias, 0.f);
        out[(size_t)(b0 + v + 8 * hi) * Psz + (p0 + ln)] = o;
    }
}

// ---------------------------------------------------------------------------
extern "C" void kernel_launch(void* const* d_in, const int* in_sizes, int n_in,
                              void* d_out, int out_size, void* d_ws, size_t ws_size,
                              hipStream_t stream) {
    const float* x    = (const float*)d_in[0];   // (B,C,H,W) == (B,C,T)
    const float* prot = (const float*)d_in[1];   // (P,C,T)
    const float* fw   = (const float*)d_in[2];   // (1,C)
    const float* fb   = (const float*)d_in[3];   // (1,)
    float* out        = (float*)d_out;           // (B,P)

    char* ws = (char*)d_ws;
    const size_t xT_bytes = (size_t)Tsz * Bsz * Csz * 2;   // 14.68 MB
    const size_t pT_bytes = (size_t)Tsz * Psz * Csz * 2;   // 29.36 MB
    __bf16* xT = (__bf16*)ws;            ws += xT_bytes;
    __bf16* yT = (__bf16*)ws;            ws += xT_bytes;
    __bf16* pT = (__bf16*)ws;            ws += pT_bytes;
    float*  xnT = (float*)ws;            ws += (size_t)Tsz * Bsz * 4;
    float*  pnT = (float*)ws;            ws += (size_t)Tsz * Psz * 4;

    prep_kernel<<<Bsz, 256, 0, stream>>>(x,    xT, yT,      fw,      xnT, Bsz);
    prep_kernel<<<Psz, 256, 0, stream>>>(prot, pT, nullptr, nullptr, pnT, Psz);

    dim3 grid(Bsz / 16, Psz / 128);      // 16 x 4 blocks, 8 waves each
    proto_wmma_kernel<<<grid, 256, 0, stream>>>(xT, yT, pT, xnT, pnT, fb, out);
}
